// PhiMoESparseMoE_89481348645454
// MI455X (gfx1250) — compile-verified, matrix operand order checked by reading, not run on hardware
//
#include <hip/hip_runtime.h>
#include <hip/hip_bf16.h>

// Problem constants (match the reference)
#define TD   1024   // hidden D
#define TF   4096   // intermediate F
#define TE   16     // experts
#define TT   2048   // tokens (B*L)
#define NSLOT (TT*2) // token-expert slots (top-2)

typedef __attribute__((ext_vector_type(16))) __bf16 v16bf;
typedef __attribute__((ext_vector_type(8)))  float  v8f;

union Tile16 {
    v16bf v;
    uint4 q[2];
    unsigned short us[16];
};

// Convert 8 fp32 -> bf16 lanes [base, base+8) of a v16bf
__device__ __forceinline__ void cvtf8(const float4 a, const float4 b, Tile16& t, int base) {
    t.v[base + 0] = (__bf16)a.x; t.v[base + 1] = (__bf16)a.y;
    t.v[base + 2] = (__bf16)a.z; t.v[base + 3] = (__bf16)a.w;
    t.v[base + 4] = (__bf16)b.x; t.v[base + 5] = (__bf16)b.y;
    t.v[base + 6] = (__bf16)b.z; t.v[base + 7] = (__bf16)b.w;
}

// Load one 16x32 bf16 operand fragment for this lane from an fp32 row.
// ISA layout (16-bit A 16x32): lanes 0-15 hold K {kt..kt+7, kt+16..kt+23},
// lanes 16-31 hold K {kt+8..kt+15, kt+24..kt+31}. khalf = lane>>4.
__device__ __forceinline__ void load_b_f32(const float* __restrict__ row, int kt, int khalf, Tile16& t) {
    const float4* p0 = (const float4*)(row + kt + khalf * 8);
    cvtf8(p0[0], p0[1], t, 0);
    const float4* p1 = (const float4*)(row + kt + 16 + khalf * 8);
    cvtf8(p1[0], p1[1], t, 8);
}

__device__ __forceinline__ void load_a_bf16(const unsigned short* __restrict__ row, int kt, int khalf, Tile16& t) {
    t.q[0] = *(const uint4*)(row + kt + khalf * 8);
    t.q[1] = *(const uint4*)(row + kt + 16 + khalf * 8);
}

__device__ __forceinline__ unsigned short f32_to_bf16_bits(float f) {
    __bf16 b = (__bf16)f;
    unsigned short u;
    __builtin_memcpy(&u, &b, 2);
    return u;
}

// ---------------------------------------------------------------------------
// Router: logits = x @ gate_w^T ; top-2 + softmax-of-2 ; build expert lists.
// block = 256 threads = 16 tokens x 16 experts.
// ---------------------------------------------------------------------------
__global__ __launch_bounds__(256) void router_kernel(
    const float* __restrict__ x, const float* __restrict__ gw,
    float* __restrict__ logits_out, float* __restrict__ topkw,
    int* __restrict__ counts, int* __restrict__ list)
{
    __shared__ float lg[16][17];
    const int e  = threadIdx.x & 15;
    const int tl = threadIdx.x >> 4;
    const int tok = blockIdx.x * 16 + tl;

    const float* xr = x + (size_t)tok * TD;
    const float* gr = gw + (size_t)e * TD;
    float acc = 0.f;
#pragma unroll 4
    for (int d = 0; d < TD; d += 4) {
        float4 xv = *(const float4*)(xr + d);
        float4 gv = *(const float4*)(gr + d);
        acc += xv.x * gv.x + xv.y * gv.y + xv.z * gv.z + xv.w * gv.w;
    }
    lg[tl][e] = acc;
    logits_out[(size_t)tok * TE + e] = acc;   // second output of the tuple
    __syncthreads();

    if (threadIdx.x < 16) {
        const int t2 = threadIdx.x;
        const int tok2 = blockIdx.x * 16 + t2;
        float best = -3.4e38f, second = -3.4e38f;
        int bi = 0, si = 0;
#pragma unroll
        for (int k = 0; k < TE; ++k) {
            float v = lg[t2][k];
            if (v > best)        { second = best; si = bi; best = v; bi = k; }
            else if (v > second) { second = v; si = k; }
        }
        float w0 = 1.f / (1.f + __expf(second - best)); // softmax over 2
        float w1 = 1.f - w0;
        topkw[tok2 * 2 + 0] = w0;
        topkw[tok2 * 2 + 1] = w1;
        int p0 = atomicAdd(&counts[bi], 1);
        list[bi * TT + p0] = tok2 * 2;       // slot encodes token*2 + k
        int p1 = atomicAdd(&counts[si], 1);
        list[si * TT + p1] = tok2 * 2 + 1;
    }
}

// ---------------------------------------------------------------------------
// x (fp32) -> bf16, 4 elements per thread.
// ---------------------------------------------------------------------------
__global__ __launch_bounds__(256) void cvt_x_kernel(
    const float* __restrict__ x, unsigned short* __restrict__ xb)
{
    int i = (blockIdx.x * blockDim.x + threadIdx.x) * 4;
    if (i < TT * TD) {
        float4 v = *(const float4*)(x + i);
        ushort4 o;
        o.x = f32_to_bf16_bits(v.x);
        o.y = f32_to_bf16_bits(v.y);
        o.z = f32_to_bf16_bits(v.z);
        o.w = f32_to_bf16_bits(v.w);
        *(ushort4*)(xb + i) = o;
    }
}

// ---------------------------------------------------------------------------
// GEMM1: H[slot, f] = silu(x . w1[f]) * (x . w3[f]) for gathered rows of expert e.
// One wave -> M=64 x N=16 tile (4 A-tiles reuse each converted B fragment).
// block = 4 waves covering 64 f-columns. grid = (F/64, maxRowBlocks=32, E).
// ---------------------------------------------------------------------------
__global__ __launch_bounds__(128) void moe_gemm13(
    const unsigned short* __restrict__ xb, const float* __restrict__ w1s,
    const float* __restrict__ w3s, const int* __restrict__ counts,
    const int* __restrict__ list, unsigned short* __restrict__ hbuf)
{
    const int e = blockIdx.z;
    const int cnt = counts[e];
    const int rowblock = blockIdx.y;
    if (rowblock * 64 >= cnt) return;

    const int wave  = threadIdx.x >> 5;
    const int lane  = threadIdx.x & 31;
    const int n     = lane & 15;
    const int khalf = lane >> 4;
    const int fbase = blockIdx.x * 64 + wave * 16;

    const int* le = list + e * TT;
    const unsigned short* arow[4];
#pragma unroll
    for (int t = 0; t < 4; ++t) {
        int ra = rowblock * 64 + t * 16 + n;
        if (ra >= cnt) ra = cnt - 1;             // clamp; masked at store
        arow[t] = xb + (size_t)(le[ra] >> 1) * TD;
    }
    const float* w1r = w1s + ((size_t)e * TF + (size_t)(fbase + n)) * TD;
    const float* w3r = w3s + ((size_t)e * TF + (size_t)(fbase + n)) * TD;

    v8f acc1[4] = {{}, {}, {}, {}};
    v8f acc3[4] = {{}, {}, {}, {}};
#pragma unroll 2
    for (int kt = 0; kt < TD; kt += 32) {
        Tile16 b1, b3;
        load_b_f32(w1r, kt, khalf, b1);
        load_b_f32(w3r, kt, khalf, b3);
#pragma unroll
        for (int t = 0; t < 4; ++t) {
            Tile16 a;
            load_a_bf16(arow[t], kt, khalf, a);
            acc1[t] = __builtin_amdgcn_wmma_f32_16x16x32_bf16(false, a.v, false, b1.v, (short)0, acc1[t], false, false);
            acc3[t] = __builtin_amdgcn_wmma_f32_16x16x32_bf16(false, a.v, false, b3.v, (short)0, acc3[t], false, false);
        }
    }

    // C layout: element i of v8f in lane l -> M = i + 8*(l>>4), N = l&15
#pragma unroll
    for (int t = 0; t < 4; ++t) {
#pragma unroll
        for (int i = 0; i < 8; ++i) {
            int m = t * 16 + i + 8 * khalf;
            int rr = rowblock * 64 + m;
            if (rr < cnt) {
                int slot = le[rr];
                float g = acc1[t][i];
                float h = (g / (1.f + __expf(-g))) * acc3[t][i];  // silu(g) * v
                hbuf[(size_t)slot * TF + fbase + n] = f32_to_bf16_bits(h);
            }
        }
    }
}

// ---------------------------------------------------------------------------
// GEMM2: out[tok, d] += cw * (H[slot] . w2[d]) for gathered rows of expert e.
// One wave -> M=64 x N=16. grid = (D/64, 32, E). Exactly 2 atomic float adds
// per out element (commutative => deterministic given zero-initialized out).
// ---------------------------------------------------------------------------
__global__ __launch_bounds__(128) void moe_gemm2(
    const unsigned short* __restrict__ hbuf, const float* __restrict__ w2s,
    const int* __restrict__ counts, const int* __restrict__ list,
    const float* __restrict__ topkw, float* __restrict__ out)
{
    const int e = blockIdx.z;
    const int cnt = counts[e];
    const int rowblock = blockIdx.y;
    if (rowblock * 64 >= cnt) return;

    const int wave  = threadIdx.x >> 5;
    const int lane  = threadIdx.x & 31;
    const int n     = lane & 15;
    const int khalf = lane >> 4;
    const int dbase = blockIdx.x * 64 + wave * 16;

    const int* le = list + e * TT;
    const unsigned short* arow[4];
#pragma unroll
    for (int t = 0; t < 4; ++t) {
        int ra = rowblock * 64 + t * 16 + n;
        if (ra >= cnt) ra = cnt - 1;
        arow[t] = hbuf + (size_t)le[ra] * TF;
    }
    const float* w2r = w2s + ((size_t)e * TD + (size_t)(dbase + n)) * TF;

    v8f acc[4] = {{}, {}, {}, {}};
#pragma unroll 2
    for (int kt = 0; kt < TF; kt += 32) {
        Tile16 b;
        load_b_f32(w2r, kt, khalf, b);
#pragma unroll
        for (int t = 0; t < 4; ++t) {
            Tile16 a;
            load_a_bf16(arow[t], kt, khalf, a);
            acc[t] = __builtin_amdgcn_wmma_f32_16x16x32_bf16(false, a.v, false, b.v, (short)0, acc[t], false, false);
        }
    }

#pragma unroll
    for (int t = 0; t < 4; ++t) {
#pragma unroll
        for (int i = 0; i < 8; ++i) {
            int m = t * 16 + i + 8 * khalf;
            int rr = rowblock * 64 + m;
            if (rr < cnt) {
                int slot = le[rr];
                float cw = topkw[slot];
                int tok = slot >> 1;
                atomicAdd(out + (size_t)tok * TD + dbase + n, acc[t][i] * cw);
            }
        }
    }
}

// ---------------------------------------------------------------------------
extern "C" void kernel_launch(void* const* d_in, const int* in_sizes, int n_in,
                              void* d_out, int out_size, void* d_ws, size_t ws_size,
                              hipStream_t stream)
{
    const float* x   = (const float*)d_in[0];   // [T, D]
    const float* gw  = (const float*)d_in[1];   // [E, D]
    const float* w1s = (const float*)d_in[2];   // [E, F, D]
    const float* w2s = (const float*)d_in[3];   // [E, D, F]
    const float* w3s = (const float*)d_in[4];   // [E, F, D]

    float* out    = (float*)d_out;               // [T, D]
    float* logits = out + (size_t)TT * TD;       // [T, E]

    // Workspace carve-out
    char* p = (char*)d_ws;
    int*   counts = (int*)p;   p += 256;
    float* topkw  = (float*)p; p += (size_t)TT * 2 * sizeof(float);
    int*   list   = (int*)p;   p += (size_t)TE * TT * sizeof(int);
    p = (char*)(((unsigned long long)p + 255ull) & ~255ull);
    unsigned short* xb   = (unsigned short*)p; p += (size_t)TT * TD * 2;      // 4 MB
    unsigned short* hbuf = (unsigned short*)p; p += (size_t)NSLOT * TF * 2;   // 32 MB

    hipMemsetAsync(out, 0, (size_t)TT * TD * sizeof(float), stream);
    hipMemsetAsync(counts, 0, TE * sizeof(int), stream);

    router_kernel<<<TT / 16, 256, 0, stream>>>(x, gw, logits, topkw, counts, list);
    cvt_x_kernel<<<(TT * TD / 4 + 255) / 256, 256, 0, stream>>>(x, xb);
    moe_gemm13<<<dim3(TF / 64, 32, TE), 128, 0, stream>>>(xb, w1s, w3s, counts, list, hbuf);
    moe_gemm2<<<dim3(TD / 64, 32, TE), 128, 0, stream>>>(hbuf, w2s, counts, list, topkw, out);
}
